// DeepseekMLAHead_27410481283072
// MI455X (gfx1250) — matrix operand order repaired and checked by reading, not compiled, not run on hardware
//
#include <hip/hip_runtime.h>
#include <math.h>

typedef unsigned short u16;
typedef unsigned int   u32;
typedef unsigned long long u64;
typedef __attribute__((ext_vector_type(8)))  float  v8f;
typedef __attribute__((ext_vector_type(16))) __bf16 v16bf;
typedef __attribute__((ext_vector_type(8)))  u32    v8u;

#define B_       2
#define S_       2048
#define HID_     2048
#define NH_      16
#define QLORA_   682
#define QLORA_P  768       // padded to 128-multiple
#define KVLORA_  256
#define KVROW_   320       // KVLORA + ROPE
#define KVROW_P  384       // padded to 128-multiple
#define THETA_   128000.0f
#define SCALE_   0.08838834764831845f   // 1/sqrt(128)
#define EPS_     1e-6f
#define ROWS_    (B_*S_)   // 4096

static __device__ __forceinline__ u16 f2bf(float f) {
  u32 u = __float_as_uint(f);
  u32 r = (u + 0x7FFFu + ((u >> 16) & 1u)) >> 16;
  return (u16)r;
}
static __device__ __forceinline__ v16bf asbf(v8u x) { return __builtin_bit_cast(v16bf, x); }
// A-matrix 16x32 bf16 per-lane K map (ISA 7.12.2)
static __device__ __forceinline__ int kmapA(int c, int lhi) {
  return (c < 4) ? (lhi * 8 + 2 * c) : (16 + lhi * 8 + 2 * (c - 4));
}
// B-matrix 32x16 bf16 per-lane K map
static __device__ __forceinline__ int kmapB(int c, int lhi) {
  return (c < 4) ? (lhi * 8 + 2 * c) : (16 + lhi * 8 + 2 * (c - 4));
}
static __device__ __forceinline__ v8f vzero() {
  v8f z;
#pragma unroll
  for (int i = 0; i < 8; i++) z[i] = 0.0f;
  return z;
}
// CDNA5 async global->LDS copy (16B per lane), tracked by ASYNCcnt
static __device__ __forceinline__ void async_ld_b128(const void* gaddr, void* lds) {
  u32 l = (u32)(size_t)lds;
  u64 g = (u64)(size_t)gaddr;
  asm volatile("global_load_async_to_lds_b128 %0, %1, off" :: "v"(l), "v"(g) : "memory");
}
static __device__ __forceinline__ void wait_async0() {
  asm volatile("s_wait_asynccnt 0x0" ::: "memory");
}

// ---------------- f32 -> bf16 convert with zero padding ----------------
__global__ __launch_bounds__(256) void convert_pad_kernel(
    const float* __restrict__ src, u16* __restrict__ dst,
    int srcR, int srcC, int dstR, int dstC) {
  size_t i = (size_t)blockIdx.x * 256 + threadIdx.x;
  size_t tot = (size_t)dstR * dstC;
  if (i >= tot) return;
  int r = (int)(i / dstC), c = (int)(i % dstC);
  float v = (r < srcR && c < srcC) ? src[(size_t)r * srcC + c] : 0.0f;
  dst[i] = f2bf(v);
}

// ---------------- RMSNorm (one row per block), bf16 out w/ pad ----------------
__global__ __launch_bounds__(256) void rmsnorm_kernel(
    const float* __restrict__ in, const float* __restrict__ w,
    u16* __restrict__ out, int cols, int ld_in, int ld_out, float inv_n) {
  __shared__ float red[256];
  int row = blockIdx.x, t = threadIdx.x;
  const float* rp = in + (size_t)row * ld_in;
  float ss = 0.0f;
  for (int c = t; c < cols; c += 256) { float v = rp[c]; ss += v * v; }
  red[t] = ss; __syncthreads();
  for (int o = 128; o > 0; o >>= 1) { if (t < o) red[t] += red[t + o]; __syncthreads(); }
  float rr = rsqrtf(red[0] * inv_n + EPS_);
  u16* op = out + (size_t)row * ld_out;
  for (int c = t; c < ld_out; c += 256)
    op[c] = (c < cols) ? f2bf(rp[c] * rr * w[c]) : (u16)0;
}

// ---------------- bf16 GEMM: C[M,N] = A[M,K]*B[K,N], f32 out ----------------
// M%128==0, N%128==0, K%32==0. 256 thr = 8 waves, 128x128 tile, double-buffered
// LDS, async A staging, reg-pipelined transposed B staging.
__global__ __launch_bounds__(256) void gemm_bf16_kernel(
    const u16* __restrict__ A, const u16* __restrict__ Bm, float* __restrict__ C,
    int N, int K) {
  const int LDT = 40;  // u16 stride: 80B rows -> 16B-aligned segments
  __shared__ __align__(16) u16 As[2][128 * 40];   // [row][k]
  __shared__ __align__(16) u16 Bs[2][128 * 40];   // transposed [n][k]
  int tid = threadIdx.x;
  int wave = tid >> 5, lane = tid & 31;
  int lr = lane & 15, lhi = lane >> 4;
  int wm = (wave >> 1) * 32, wn = (wave & 1) * 64;
  int bm = blockIdx.y * 128, bn = blockIdx.x * 128;
  int nkt = K >> 5;
  v8f acc[2][4];
#pragma unroll
  for (int i = 0; i < 2; i++)
#pragma unroll
    for (int j = 0; j < 4; j++) acc[i][j] = vzero();

  u32 br[8];
  // ---- prologue: stage tile 0 ----
#pragma unroll
  for (int i = 0; i < 2; i++) {            // async A: 512 x 16B chunks
    int p = tid + i * 256;
    int r = p >> 2, seg = p & 3;
    async_ld_b128(A + (size_t)(bm + r) * K + seg * 8, &As[0][r * LDT + seg * 8]);
  }
#pragma unroll
  for (int i = 0; i < 8; i++) {            // B: 2048 u32 pairs
    int p = tid + i * 256;
    int rb = p >> 6, cpb = p & 63;
    br[i] = *(const u32*)(Bm + (size_t)rb * N + bn + cpb * 2);
  }
#pragma unroll
  for (int i = 0; i < 8; i++) {
    int p = tid + i * 256;
    int rb = p >> 6, cpb = p & 63;
    Bs[0][(cpb * 2) * LDT + rb]     = (u16)(br[i] & 0xFFFFu);
    Bs[0][(cpb * 2 + 1) * LDT + rb] = (u16)(br[i] >> 16);
  }
  wait_async0();
  __syncthreads();

  for (int kt = 0; kt < nkt; kt++) {
    int cur = kt & 1, nxt = cur ^ 1;
    bool more = (kt + 1 < nkt);
    int k0n = (kt + 1) << 5;
    if (more) {
      // async A for next tile (As[nxt] free: last read in the previous iteration)
#pragma unroll
      for (int i = 0; i < 2; i++) {
        int p = tid + i * 256;
        int r = p >> 2, seg = p & 3;
        async_ld_b128(A + (size_t)(bm + r) * K + k0n + seg * 8,
                      &As[nxt][r * LDT + seg * 8]);
      }
#pragma unroll
      for (int i = 0; i < 8; i++) {
        int p = tid + i * 256;
        int rb = p >> 6, cpb = p & 63;
        br[i] = *(const u32*)(Bm + (size_t)(k0n + rb) * N + bn + cpb * 2);
      }
      if (kt + 2 < nkt) {
        int k2 = (kt + 2) << 5;
        __builtin_prefetch(A + (size_t)(bm + (tid >> 1)) * K + k2 + (tid & 1) * 16, 0, 1);
        __builtin_prefetch(Bm + (size_t)(k2 + (tid >> 3)) * N + bn + (tid & 7) * 16, 0, 1);
      }
    }
    // ---- compute from LDS[cur] ----
    const u16* asb = As[cur];
    const u16* bsb = Bs[cur];
    v8u afr[2], bfr[4];
#pragma unroll
    for (int f = 0; f < 2; f++) {
      const u16* ap = asb + (wm + f * 16 + lr) * LDT;
      v8u a;
#pragma unroll
      for (int c = 0; c < 8; c++) a[c] = *(const u32*)(ap + kmapA(c, lhi));
      afr[f] = a;
    }
#pragma unroll
    for (int f = 0; f < 4; f++) {
      const u16* bp = bsb + (wn + f * 16 + lr) * LDT;
      v8u b;
#pragma unroll
      for (int c = 0; c < 8; c++) b[c] = *(const u32*)(bp + kmapB(c, lhi));
      bfr[f] = b;
    }
#pragma unroll
    for (int i = 0; i < 2; i++)
#pragma unroll
      for (int j = 0; j < 4; j++)
        acc[i][j] = __builtin_amdgcn_wmma_f32_16x16x32_bf16(
            false, asbf(afr[i]), false, asbf(bfr[j]),
            (short)0, acc[i][j], false, false);
    if (more) {
      __syncthreads();                     // everyone done reading Bs[cur]
#pragma unroll
      for (int i = 0; i < 8; i++) {
        int p = tid + i * 256;
        int rb = p >> 6, cpb = p & 63;
        Bs[nxt][(cpb * 2) * LDT + rb]     = (u16)(br[i] & 0xFFFFu);
        Bs[nxt][(cpb * 2 + 1) * LDT + rb] = (u16)(br[i] >> 16);
      }
      wait_async0();                       // A[nxt] landed
      __syncthreads();
    }
  }
  // ---- epilogue: incremental row pointers ----
#pragma unroll
  for (int i = 0; i < 2; i++) {
#pragma unroll
    for (int j = 0; j < 4; j++) {
      float* cp0 = C + (size_t)(bm + wm + i * 16 + 8 * lhi) * N + (bn + wn + j * 16 + lr);
#pragma unroll
      for (int r = 0; r < 8; r++) { *cp0 = acc[i][j][r]; cp0 += N; }
    }
  }
}

// ---------------- assemble qf/kf/v (bf16, [B,NH,S,128]) + RoPE ----------------
__global__ __launch_bounds__(128) void assemble_kernel(
    const float* __restrict__ q, const float* __restrict__ kv,
    const float* __restrict__ ckv_kpe,
    u16* __restrict__ qf, u16* __restrict__ kf, u16* __restrict__ vv) {
  int s = blockIdx.x, h = blockIdx.y, b = blockIdx.z;
  int t = threadIdx.x;
  size_t rowbs = (size_t)b * S_ + s;
  const float* qrow  = q + rowbs * 2048 + h * 128;
  const float* kvrow = kv + rowbs * 3072 + h * 192;
  const float* kpe   = ckv_kpe + rowbs * KVROW_P + KVLORA_;
  size_t ob = ((size_t)(b * NH_ + h) * S_ + s) * 128;
  if (t < 64) { qf[ob + t] = f2bf(qrow[t]); kf[ob + t] = f2bf(kvrow[t]); }
  vv[ob + t] = f2bf(kvrow[64 + t]);
  if (t < 32) {
    float inv = powf(THETA_, -(float)t * (1.0f / 32.0f));
    float ang = (float)s * inv;
    float cs = cosf(ang), sn = sinf(ang);
    float x0 = qrow[64 + 2 * t], x1 = qrow[64 + 2 * t + 1];
    qf[ob + 64 + t] = f2bf(x0 * cs - x1 * sn);
    qf[ob + 96 + t] = f2bf(x1 * cs + x0 * sn);
    float k0 = kpe[2 * t], k1 = kpe[2 * t + 1];
    kf[ob + 64 + t] = f2bf(k0 * cs - k1 * sn);
    kf[ob + 96 + t] = f2bf(k1 * cs + k0 * sn);
  }
}

// ---------------- flash attention: 4 waves share K/V tiles, 64 q rows/block ----------------
__global__ __launch_bounds__(128) void mla_attn_kernel(
    const u16* __restrict__ qf, const u16* __restrict__ kf,
    const u16* __restrict__ vv, u16* __restrict__ outp) {
  __shared__ __align__(16) u16 ks[32 * 136];  // [key][d], 272B rows (16B aligned)
  __shared__ u16 vt[128 * 34];                // [d][key]
  __shared__ u16 ps[4][16 * 34];              // per-wave P tile
  int tid = threadIdx.x;
  int wave = tid >> 5, lane = tid & 31;
  int lr = lane & 15, lhi = lane >> 4;
  int q0 = blockIdx.x * 64;
  int q0w = q0 + wave * 16;
  int bh = blockIdx.y;
  int b = bh / NH_, h = bh % NH_;
  const u16* qb = qf + (size_t)bh * S_ * 128;
  const u16* kb = kf + (size_t)bh * S_ * 128;
  const u16* vb = vv + (size_t)bh * S_ * 128;

  v8u aq[4];
  {
    const u16* qrow = qb + (size_t)(q0w + lr) * 128;
#pragma unroll
    for (int d0 = 0; d0 < 4; d0++) {
      v8u a;
#pragma unroll
      for (int c = 0; c < 8; c++) a[c] = *(const u32*)(qrow + d0 * 32 + kmapA(c, lhi));
      aq[d0] = a;
    }
  }
  float m_i[8], l_i[8];
  v8f o_acc[8];
#pragma unroll
  for (int r = 0; r < 8; r++) { m_i[r] = -3.0e38f; l_i[r] = 0.0f; }
#pragma unroll
  for (int f = 0; f < 8; f++) o_acc[f] = vzero();

  int nkb = (q0 + 64 + 31) / 32;
  for (int jb = 0; jb < nkb; jb++) {
    int kbb = jb * 32;
    // async K tile: 128 x 16B chunks
#pragma unroll
    for (int i = 0; i < 1; i++) {
      int p = tid;                       // 0..127 -> row 0..31, seg 0..3
      int row = p >> 2, seg = p & 3;
      async_ld_b128(kb + (size_t)(kbb + row) * 128 + seg * 8,
                    &ks[row * 136 + seg * 8]);
    }
    // remaining 3/4 of K tile (512 chunks total? no: 32*128*2B = 8KB = 512x16B/ lane count)
    {
      int p = tid + 128;
      int row = p >> 2, seg = p & 3;
      async_ld_b128(kb + (size_t)(kbb + row) * 128 + seg * 8,
                    &ks[row * 136 + seg * 8]);
      p = tid + 256;
      row = p >> 2; seg = p & 3;
      async_ld_b128(kb + (size_t)(kbb + row) * 128 + seg * 8,
                    &ks[row * 136 + seg * 8]);
      p = tid + 384;
      row = p >> 2; seg = p & 3;
      async_ld_b128(kb + (size_t)(kbb + row) * 128 + seg * 8,
                    &ks[row * 136 + seg * 8]);
    }
    // V tile transposed (manual)
    for (int idx = tid; idx < 2048; idx += 128) {
      int row = idx >> 6, cp = idx & 63;
      u32 vval = *(const u32*)(vb + (size_t)(kbb + row) * 128 + cp * 2);
      vt[(cp * 2) * 34 + row]     = (u16)(vval & 0xFFFFu);
      vt[(cp * 2 + 1) * 34 + row] = (u16)(vval >> 16);
    }
    wait_async0();
    __syncthreads();
    // QK^T from LDS
    v8f sc[2];
#pragma unroll
    for (int nf = 0; nf < 2; nf++) {
      const u16* krow = ks + (nf * 16 + lr) * 136;
      v8f c = vzero();
#pragma unroll
      for (int d0 = 0; d0 < 4; d0++) {
        v8u bfr;
#pragma unroll
        for (int ch = 0; ch < 8; ch++) bfr[ch] = *(const u32*)(krow + d0 * 32 + kmapB(ch, lhi));
        c = __builtin_amdgcn_wmma_f32_16x16x32_bf16(
            false, asbf(aq[d0]), false, asbf(bfr), (short)0, c, false, false);
      }
      sc[nf] = c;
    }
    // online softmax
    float alpha[8];
#pragma unroll
    for (int r = 0; r < 8; r++) {
      int row = q0w + r + 8 * lhi;
      float v0 = sc[0][r] * SCALE_;
      float v1 = sc[1][r] * SCALE_;
      if (kbb + lr > row)      v0 = -1.0e9f;
      if (kbb + 16 + lr > row) v1 = -1.0e9f;
      float mx = fmaxf(v0, v1);
#pragma unroll
      for (int off = 1; off < 16; off <<= 1) mx = fmaxf(mx, __shfl_xor(mx, off, 32));
      float mn = fmaxf(m_i[r], mx);
      float a  = __expf(m_i[r] - mn);
      float p0 = __expf(v0 - mn);
      float p1 = __expf(v1 - mn);
      float rs = p0 + p1;
#pragma unroll
      for (int off = 1; off < 16; off <<= 1) rs += __shfl_xor(rs, off, 32);
      l_i[r] = l_i[r] * a + rs;
      m_i[r] = mn;
      alpha[r] = a;
      ps[wave][(r + 8 * lhi) * 34 + lr]      = f2bf(p0);
      ps[wave][(r + 8 * lhi) * 34 + 16 + lr] = f2bf(p1);
    }
#pragma unroll
    for (int f = 0; f < 8; f++)
#pragma unroll
      for (int r = 0; r < 8; r++) o_acc[f][r] *= alpha[r];
    __syncthreads();   // ps visible (also orders own-wave DS ops)
    // P*V
    v8u pa;
#pragma unroll
    for (int c = 0; c < 8; c++) pa[c] = *(const u32*)(&ps[wave][lr * 34 + kmapA(c, lhi)]);
#pragma unroll
    for (int nf = 0; nf < 8; nf++) {
      v8u bfr;
#pragma unroll
      for (int c = 0; c < 8; c++) bfr[c] = *(const u32*)(vt + (nf * 16 + lr) * 34 + kmapB(c, lhi));
      o_acc[nf] = __builtin_amdgcn_wmma_f32_16x16x32_bf16(
          false, asbf(pa), false, asbf(bfr), (short)0, o_acc[nf], false, false);
    }
    __syncthreads();   // before next tile overwrites ks/vt
  }
  // write out bf16: [b*S+row][h*128 + d]
#pragma unroll
  for (int nf = 0; nf < 8; nf++) {
#pragma unroll
    for (int r = 0; r < 8; r++) {
      int row = q0w + r + 8 * lhi;
      float val = o_acc[nf][r] / l_i[r];
      outp[((size_t)b * S_ + row) * 2048 + h * 128 + nf * 16 + lr] = f2bf(val);
    }
  }
}

// ---------------- host ----------------
extern "C" void kernel_launch(void* const* d_in, const int* in_sizes, int n_in,
                              void* d_out, int out_size, void* d_ws, size_t ws_size,
                              hipStream_t stream) {
  const float* x         = (const float*)d_in[0];
  const float* q_a_w     = (const float*)d_in[1];
  const float* q_a_ln_w  = (const float*)d_in[2];
  const float* q_b_w     = (const float*)d_in[3];
  const float* kv_a_w    = (const float*)d_in[4];
  const float* kv_a_ln_w = (const float*)d_in[5];
  const float* kv_b_w    = (const float*)d_in[6];
  const float* o_w       = (const float*)d_in[7];
  float* out = (float*)d_out;
  (void)in_sizes; (void)n_in; (void)out_size; (void)ws_size;

  char* ws = (char*)d_ws;
  size_t off = 0;
  auto alloc = [&](size_t bytes) { void* p = ws + off; off = (off + bytes + 255) & ~(size_t)255; return p; };

  u16*   x_bf    = (u16*)  alloc((size_t)ROWS_ * HID_ * 2);
  u16*   qaw_bf  = (u16*)  alloc((size_t)HID_ * QLORA_P * 2);
  u16*   qbw_bf  = (u16*)  alloc((size_t)QLORA_P * 2048 * 2);
  u16*   kvaw_bf = (u16*)  alloc((size_t)HID_ * KVROW_P * 2);
  u16*   kvbw_bf = (u16*)  alloc((size_t)KVLORA_ * 3072 * 2);
  u16*   ow_bf   = (u16*)  alloc((size_t)2048 * 2048 * 2);
  float* qa_f    = (float*)alloc((size_t)ROWS_ * QLORA_P * 4);
  u16*   qan_bf  = (u16*)  alloc((size_t)ROWS_ * QLORA_P * 2);
  float* ckv_f   = (float*)alloc((size_t)ROWS_ * KVROW_P * 4);
  u16*   ckvn_bf = (u16*)  alloc((size_t)ROWS_ * KVLORA_ * 2);
  float* q_f     = (float*)alloc((size_t)ROWS_ * 2048 * 4);
  float* kv_f    = (float*)alloc((size_t)ROWS_ * 3072 * 4);
  u16*   qf_bf   = (u16*)  alloc((size_t)B_ * NH_ * S_ * 128 * 2);
  u16*   kf_bf   = (u16*)  alloc((size_t)B_ * NH_ * S_ * 128 * 2);
  u16*   v_bf    = (u16*)  alloc((size_t)B_ * NH_ * S_ * 128 * 2);
  u16*   attn_bf = (u16*)  alloc((size_t)ROWS_ * 2048 * 2);

  auto cvt = [&](const float* src, u16* dst, int sR, int sC, int dR, int dC) {
    size_t tot = (size_t)dR * dC;
    convert_pad_kernel<<<dim3((unsigned)((tot + 255) / 256)), dim3(256), 0, stream>>>(src, dst, sR, sC, dR, dC);
  };
  cvt(x,      x_bf,    ROWS_,   HID_,   ROWS_,   HID_);
  cvt(q_a_w,  qaw_bf,  HID_,    QLORA_, HID_,    QLORA_P);
  cvt(q_b_w,  qbw_bf,  QLORA_,  2048,   QLORA_P, 2048);
  cvt(kv_a_w, kvaw_bf, HID_,    KVROW_, HID_,    KVROW_P);
  cvt(kv_b_w, kvbw_bf, KVLORA_, 3072,   KVLORA_, 3072);
  cvt(o_w,    ow_bf,   2048,    2048,   2048,    2048);

  auto gemm = [&](const u16* A, const u16* Bm, float* C, int N, int K) {
    gemm_bf16_kernel<<<dim3(N / 128, ROWS_ / 128), dim3(256), 0, stream>>>(A, Bm, C, N, K);
  };
  gemm(x_bf, qaw_bf,  qa_f,  QLORA_P, HID_);
  gemm(x_bf, kvaw_bf, ckv_f, KVROW_P, HID_);

  rmsnorm_kernel<<<ROWS_, 256, 0, stream>>>(qa_f,  q_a_ln_w,  qan_bf,  QLORA_,  QLORA_P, QLORA_P, 1.0f / QLORA_);
  rmsnorm_kernel<<<ROWS_, 256, 0, stream>>>(ckv_f, kv_a_ln_w, ckvn_bf, KVLORA_, KVROW_P, KVLORA_, 1.0f / KVLORA_);

  gemm(qan_bf,  qbw_bf,  q_f,  2048, QLORA_P);
  gemm(ckvn_bf, kvbw_bf, kv_f, 3072, KVLORA_);

  assemble_kernel<<<dim3(S_, NH_, B_), dim3(128), 0, stream>>>(q_f, kv_f, ckv_f, qf_bf, kf_bf, v_bf);
  mla_attn_kernel<<<dim3(S_ / 64, B_ * NH_), dim3(128), 0, stream>>>(qf_bf, kf_bf, v_bf, attn_bf);

  gemm(attn_bf, ow_bf, out, 2048, HID_);
}
